// mySAGE_87703232184763
// MI455X (gfx1250) — compile-verified, matrix operand order checked by reading, not run on hardware
//
#include <hip/hip_runtime.h>
#include <hip/hip_bf16.h>
#include <cstdint>
#include <cstddef>

// ---------------- CDNA5 WMMA / TDM types ----------------
typedef __bf16 bf16_t;
typedef __attribute__((ext_vector_type(16))) __bf16 v16bf;
typedef __attribute__((ext_vector_type(8)))  float  v8f;
typedef __attribute__((ext_vector_type(4)))  unsigned u32x4;
typedef __attribute__((ext_vector_type(8)))  unsigned u32x8;

#define N_NODES   50000
#define NEIGH     12
#define H         128          // hidden width (bond and angle branches)
#define KB1       480          // bond layer-1 K  (40 gbf * 12 neigh)
#define KA1       1152         // angle layer-1 K (8 gbf * 144 pairs)
#define N_CRYS    500
#define HCAT      256
#define MTILES    (N_NODES / 16)   // 3125, exact

// f32 -> bf16 round-to-nearest-even via bits (avoids any cast portability issue)
__device__ __forceinline__ bf16_t f32_to_bf16(float f) {
  union { float f; unsigned u; } v; v.f = f;
  unsigned r = v.u + 0x7FFFu + ((v.u >> 16) & 1u);
  unsigned short h = (unsigned short)(r >> 16);
  bf16_t out; __builtin_memcpy(&out, &h, 2);
  return out;
}
__device__ __forceinline__ float bf16bits_to_f32(unsigned short h) {
  union { unsigned u; float f; } v; v.u = ((unsigned)h) << 16; return v.f;
}

// ---------------- small utility kernels ----------------
__global__ void k_cvt_bf16(const float* __restrict__ in, bf16_t* __restrict__ out, int n) {
  int i = blockIdx.x * blockDim.x + threadIdx.x;
  if (i < n) out[i] = f32_to_bf16(in[i]);
}

__global__ void k_zero_f32(float* __restrict__ p, int n) {
  int i = blockIdx.x * blockDim.x + threadIdx.x;
  if (i < n) p[i] = 0.0f;
}

// count incoming edges per node (dst = nbr_idx.flatten())
__global__ void k_count(const int* __restrict__ nbr, float* __restrict__ cnt, int nedges) {
  int e = blockIdx.x * blockDim.x + threadIdx.x;
  if (e < nedges) atomicAdd(&cnt[nbr[e]], 1.0f);
}

// scatter: agg[nbr[i][j]] += yl[i]   (message from src=i to dst=nbr)
__global__ __launch_bounds__(H) void k_scatter(const float* __restrict__ yl,
                                               const int* __restrict__ nbr,
                                               float* __restrict__ agg) {
  int i = blockIdx.x, c = threadIdx.x;           // blockDim.x == H
  float v = yl[(size_t)i * H + c];
  #pragma unroll
  for (int j = 0; j < NEIGH; ++j) {
    int d = nbr[i * NEIGH + j];
    atomicAdd(&agg[(size_t)d * H + c], v);
  }
}

// out = relu(agg/max(cnt,1) + bias + yr), stored bf16 for the next WMMA layer
__global__ void k_combine(const float* __restrict__ agg, const float* __restrict__ cnt,
                          const float* __restrict__ yr, const float* __restrict__ bias,
                          bf16_t* __restrict__ xout, int total) {
  int idx = blockIdx.x * blockDim.x + threadIdx.x;
  if (idx >= total) return;
  int i = idx >> 7, c = idx & (H - 1);
  float v = agg[idx] / fmaxf(cnt[i], 1.0f) + bias[c] + yr[idx];
  xout[idx] = f32_to_bf16(fmaxf(v, 0.0f));
}

// ---------------- WMMA fragment store (16x16 f32 C/D layout, ISA 7.12.2) ----------------
__device__ __forceinline__ void store_cd(float* __restrict__ dst, int m0, int n0,
                                         int lane, const v8f& acc, int rowstride) {
  int n = n0 + (lane & 15);
  int mrow = (lane >> 4) * 8;
  #pragma unroll
  for (int v = 0; v < 8; ++v)
    dst[(size_t)(m0 + mrow + v) * rowstride + n] = acc[v];
}

// ---------------- fused GBF-expand + dual GEMM (bond, K=480) ----------------
__global__ __launch_bounds__(256) void k_bond_gbf_gemm(
    const float* __restrict__ bond, const bf16_t* __restrict__ Wl,
    const bf16_t* __restrict__ Wr, float* __restrict__ yl, float* __restrict__ yr)
{
  __shared__ __align__(32) bf16_t As[16 * KB1];
  __shared__ float bfea[16 * NEIGH];
  const int tid = threadIdx.x;
  const int m0 = blockIdx.x * 16;

  if (tid < 16 * NEIGH) bfea[tid] = bond[(size_t)m0 * NEIGH + tid];
  __syncthreads();

  // Gaussian basis: exp(-(d - k*8/39)^2 / 0.2^2), 40 filters per neighbor
  for (int t = tid; t < 16 * KB1; t += 256) {
    int r = t / KB1, f = t - r * KB1;
    int j = f / 40, k = f - j * 40;
    float d  = bfea[r * NEIGH + j];
    float dd = d - (float)k * (8.0f / 39.0f);
    As[t] = f32_to_bf16(__expf(-dd * dd * 25.0f));
  }
  __syncthreads();

  const int wave = tid >> 5, lane = tid & 31;
  const int n0 = wave * 16;
  const int r16 = lane & 15, khalf = (lane >> 4) * 16;
  const bf16_t* ap  = As + r16 * KB1 + khalf;
  const bf16_t* wlp = Wl + (size_t)(n0 + r16) * KB1 + khalf;
  const bf16_t* wrp = Wr + (size_t)(n0 + r16) * KB1 + khalf;

  v8f accl = {0.f,0.f,0.f,0.f,0.f,0.f,0.f,0.f};
  v8f accr = {0.f,0.f,0.f,0.f,0.f,0.f,0.f,0.f};
  for (int kk = 0; kk < KB1; kk += 32) {
    __builtin_prefetch(wlp + kk + 64, 0, 1);
    v16bf a  = *(const v16bf*)(ap  + kk);
    v16bf bl = *(const v16bf*)(wlp + kk);
    v16bf br = *(const v16bf*)(wrp + kk);
    accl = __builtin_amdgcn_wmma_f32_16x16x32_bf16(false, a, false, bl, (short)0, accl, false, false);
    accr = __builtin_amdgcn_wmma_f32_16x16x32_bf16(false, a, false, br, (short)0, accr, false, false);
  }
  store_cd(yl, m0, n0, lane, accl, H);
  store_cd(yr, m0, n0, lane, accr, H);
}

// ---------------- fused GBF-expand + dual GEMM (angle, K=1152) ----------------
__global__ __launch_bounds__(256) void k_angle_gbf_gemm(
    const float* __restrict__ ang, const bf16_t* __restrict__ Wl,
    const bf16_t* __restrict__ Wr, float* __restrict__ yl, float* __restrict__ yr)
{
  __shared__ __align__(32) bf16_t As[16 * KA1];
  __shared__ float afea[16 * NEIGH * NEIGH];
  const int tid = threadIdx.x;
  const int m0 = blockIdx.x * 16;

  for (int t = tid; t < 16 * NEIGH * NEIGH; t += 256)
    afea[t] = ang[(size_t)m0 * (NEIGH * NEIGH) + t];
  __syncthreads();

  // Gaussian basis: exp(-(a - (-1 + k*2/7))^2 / 0.25^2), 8 filters per pair
  for (int t = tid; t < 16 * KA1; t += 256) {
    int r = t / KA1, f = t - r * KA1;
    int p = f >> 3, k = f & 7;
    float a  = afea[r * (NEIGH * NEIGH) + p];
    float dd = a - (-1.0f + (float)k * (2.0f / 7.0f));
    As[t] = f32_to_bf16(__expf(-dd * dd * 16.0f));
  }
  __syncthreads();

  const int wave = tid >> 5, lane = tid & 31;
  const int n0 = wave * 16;
  const int r16 = lane & 15, khalf = (lane >> 4) * 16;
  const bf16_t* ap  = As + r16 * KA1 + khalf;
  const bf16_t* wlp = Wl + (size_t)(n0 + r16) * KA1 + khalf;
  const bf16_t* wrp = Wr + (size_t)(n0 + r16) * KA1 + khalf;

  v8f accl = {0.f,0.f,0.f,0.f,0.f,0.f,0.f,0.f};
  v8f accr = {0.f,0.f,0.f,0.f,0.f,0.f,0.f,0.f};
  for (int kk = 0; kk < KA1; kk += 32) {
    __builtin_prefetch(wlp + kk + 64, 0, 1);
    v16bf a  = *(const v16bf*)(ap  + kk);
    v16bf bl = *(const v16bf*)(wlp + kk);
    v16bf br = *(const v16bf*)(wrp + kk);
    accl = __builtin_amdgcn_wmma_f32_16x16x32_bf16(false, a, false, bl, (short)0, accl, false, false);
    accr = __builtin_amdgcn_wmma_f32_16x16x32_bf16(false, a, false, br, (short)0, accr, false, false);
  }
  store_cd(yl, m0, n0, lane, accl, H);
  store_cd(yr, m0, n0, lane, accr, H);
}

// ---------------- hidden-layer dual GEMM (K=128), A tile staged via TDM ----------------
// One TENSOR_LOAD_TO_LDS per block DMAs the 16x128 bf16 A tile into LDS
// (D# group0/group1 per CDNA5 ISA), tracked by TENSORcnt.
// All 8 waves then read A fragments from LDS and stream only weights from global.
__global__ __launch_bounds__(256) void k_gemm_h128(
    const bf16_t* __restrict__ x, const bf16_t* __restrict__ Wl,
    const bf16_t* __restrict__ Wr, float* __restrict__ yl, float* __restrict__ yr)
{
  __shared__ __align__(32) bf16_t As[16 * H];
  const int tid = threadIdx.x;
  const int m0 = blockIdx.x * 16;
  const int wave = tid >> 5, lane = tid & 31;

  if (wave == 0) {
    unsigned long long ga = (unsigned long long)(const void*)(x + (size_t)m0 * H);
    unsigned lds = (unsigned)(unsigned long long)(const void*)As; // LDS aperture low bits = wave-relative offset
    u32x4 g0;
    u32x8 g1;
    g0[0] = 1u;                                            // count=1, user descriptor
    g0[1] = lds;                                           // lds_addr
    g0[2] = (unsigned)(ga & 0xFFFFFFFFull);                // global_addr[31:0]
    g0[3] = (unsigned)((ga >> 32) & 0x01FFFFFFull)         // global_addr[56:32]
            | (2u << 30);                                  // type = 2 ("image")
    g1[0] = 1u << 16;                                      // workgroup_mask=0, data_size=1 (2 bytes)
    g1[1] = (unsigned)H << 16;                             // tensor_dim0 = 128 (bits 79:48)
    g1[2] = ((unsigned)N_NODES & 0xFFFFu) << 16;           // tensor_dim1 lo (bits 95:80)
    g1[3] = ((unsigned)N_NODES >> 16)                      // tensor_dim1 hi (bits 111:96)
            | ((unsigned)H << 16);                         // tile_dim0 = 128 (bits 127:112)
    g1[4] = 16u;                                           // tile_dim1 = 16, tile_dim2 = 0
    g1[5] = (unsigned)H;                                   // tensor_dim0_stride = 128 (bits 191:160)
    g1[6] = 0u;                                            // stride hi / dim1_stride lo
    g1[7] = 0u;
    asm volatile("tensor_load_to_lds %0, %1" : : "s"(g0), "s"(g1) : "memory");
    __builtin_amdgcn_s_wait_tensorcnt(0);
  }
  __syncthreads();

  const int n0 = wave * 16;
  const int r16 = lane & 15, khalf = (lane >> 4) * 16;
  const bf16_t* ap  = As + r16 * H + khalf;
  const bf16_t* wlp = Wl + (size_t)(n0 + r16) * H + khalf;
  const bf16_t* wrp = Wr + (size_t)(n0 + r16) * H + khalf;

  v8f accl = {0.f,0.f,0.f,0.f,0.f,0.f,0.f,0.f};
  v8f accr = {0.f,0.f,0.f,0.f,0.f,0.f,0.f,0.f};
  #pragma unroll
  for (int kk = 0; kk < H; kk += 32) {
    v16bf a  = *(const v16bf*)(ap  + kk);
    v16bf bl = *(const v16bf*)(wlp + kk);
    v16bf br = *(const v16bf*)(wrp + kk);
    accl = __builtin_amdgcn_wmma_f32_16x16x32_bf16(false, a, false, bl, (short)0, accl, false, false);
    accr = __builtin_amdgcn_wmma_f32_16x16x32_bf16(false, a, false, br, (short)0, accr, false, false);
  }
  store_cd(yl, m0, n0, lane, accl, H);
  store_cd(yr, m0, n0, lane, accr, H);
}

// ---------------- segment mean pool over crystals (contiguous ranges) ----------------
__global__ __launch_bounds__(HCAT) void k_pool(const unsigned short* __restrict__ xb,
                                               const unsigned short* __restrict__ xa,
                                               const int* __restrict__ cidx,
                                               float* __restrict__ pooled) {
  int cr = blockIdx.x, c = threadIdx.x;
  int s = cidx[2 * cr], e = cidx[2 * cr + 1];
  const unsigned short* src = (c < H) ? xb : xa;
  int cc = (c < H) ? c : c - H;
  float acc = 0.0f;
  for (int r = s; r < e; ++r) acc += bf16bits_to_f32(src[(size_t)r * H + cc]);
  pooled[(size_t)cr * HCAT + c] = acc / fmaxf((float)(e - s), 1.0f);
}

// ---------------- small f32 linear: out = in @ W.T + b ----------------
__global__ void k_linear(const float* __restrict__ in, const float* __restrict__ W,
                         const float* __restrict__ b, float* __restrict__ out,
                         int rows, int K, int OUT) {
  int idx = blockIdx.x * blockDim.x + threadIdx.x;
  if (idx >= rows * OUT) return;
  int r = idx / OUT, o = idx - r * OUT;
  const float* ip = in + (size_t)r * K;
  const float* wp = W + (size_t)o * K;
  float acc = b[o];
  for (int k = 0; k < K; ++k) acc += ip[k] * wp[k];
  out[idx] = acc;
}

// ---------------- host orchestration ----------------
extern "C" void kernel_launch(void* const* d_in, const int* in_sizes, int n_in,
                              void* d_out, int out_size, void* d_ws, size_t ws_size,
                              hipStream_t stream) {
  const float* bond   = (const float*)d_in[0];
  const float* angle  = (const float*)d_in[1];
  /* d_in[2] species: unused by the reference */
  const int*   nbr    = (const int*)d_in[3];
  const int*   cidx   = (const int*)d_in[4];
  const float* Wl_b1  = (const float*)d_in[5];
  const float* Wr_b1  = (const float*)d_in[6];
  const float* b_b1   = (const float*)d_in[7];
  const float* Wl_a1  = (const float*)d_in[8];
  const float* Wr_a1  = (const float*)d_in[9];
  const float* b_a1   = (const float*)d_in[10];
  const float* Wl_b   = (const float*)d_in[11];
  const float* Wr_b   = (const float*)d_in[12];
  const float* b_b    = (const float*)d_in[13];
  const float* Wl_a   = (const float*)d_in[14];
  const float* Wr_a   = (const float*)d_in[15];
  const float* b_a    = (const float*)d_in[16];
  const float* W_mlp  = (const float*)d_in[17];
  const float* b_mlp  = (const float*)d_in[18];
  const float* W_fc   = (const float*)d_in[19];
  const float* b_fc   = (const float*)d_in[20];

  char* ws = (char*)d_ws;
  size_t off = 0;
  auto alloc = [&](size_t bytes) -> void* {
    void* p = ws + off;
    off += (bytes + 255) & ~(size_t)255;
    return p;
  };

  // bf16 weight mirrors
  bf16_t* wlb1 = (bf16_t*)alloc((size_t)H * KB1 * 2);
  bf16_t* wrb1 = (bf16_t*)alloc((size_t)H * KB1 * 2);
  bf16_t* wla1 = (bf16_t*)alloc((size_t)H * KA1 * 2);
  bf16_t* wra1 = (bf16_t*)alloc((size_t)H * KA1 * 2);
  bf16_t* wlb  = (bf16_t*)alloc((size_t)2 * H * H * 2);
  bf16_t* wrb  = (bf16_t*)alloc((size_t)2 * H * H * 2);
  bf16_t* wla  = (bf16_t*)alloc((size_t)2 * H * H * 2);
  bf16_t* wra  = (bf16_t*)alloc((size_t)2 * H * H * 2);
  // activations / scratch
  float*  yl   = (float*)alloc((size_t)N_NODES * H * 4);
  float*  yr   = (float*)alloc((size_t)N_NODES * H * 4);
  float*  agg  = (float*)alloc((size_t)N_NODES * H * 4);
  float*  cnt  = (float*)alloc((size_t)N_NODES * 4);
  bf16_t* xb   = (bf16_t*)alloc((size_t)N_NODES * H * 2);
  bf16_t* xa   = (bf16_t*)alloc((size_t)N_NODES * H * 2);
  float*  pooled = (float*)alloc((size_t)N_CRYS * HCAT * 4);
  float*  t1     = (float*)alloc((size_t)N_CRYS * HCAT * 4);
  float*  t2     = (float*)alloc((size_t)N_CRYS * HCAT * 4);

  auto cvt = [&](const float* src, bf16_t* dst, int n) {
    k_cvt_bf16<<<(n + 255) / 256, 256, 0, stream>>>(src, dst, n);
  };
  cvt(Wl_b1, wlb1, H * KB1);
  cvt(Wr_b1, wrb1, H * KB1);
  cvt(Wl_a1, wla1, H * KA1);
  cvt(Wr_a1, wra1, H * KA1);
  cvt(Wl_b,  wlb,  2 * H * H);
  cvt(Wr_b,  wrb,  2 * H * H);
  cvt(Wl_a,  wla,  2 * H * H);
  cvt(Wr_a,  wra,  2 * H * H);

  const int NH = N_NODES * H;
  const int NE = N_NODES * NEIGH;

  // edge in-degree (float), shared by all layers
  k_zero_f32<<<(N_NODES + 255) / 256, 256, 0, stream>>>(cnt, N_NODES);
  k_count<<<(NE + 255) / 256, 256, 0, stream>>>(nbr, cnt, NE);

  auto agg_combine = [&](const float* bias, bf16_t* xout) {
    k_zero_f32<<<(NH + 255) / 256, 256, 0, stream>>>(agg, NH);
    k_scatter<<<N_NODES, H, 0, stream>>>(yl, nbr, agg);
    k_combine<<<(NH + 255) / 256, 256, 0, stream>>>(agg, cnt, yr, bias, xout, NH);
  };

  // ---- bond branch ----
  k_bond_gbf_gemm<<<MTILES, 256, 0, stream>>>(bond, wlb1, wrb1, yl, yr);
  agg_combine(b_b1, xb);
  for (int l = 0; l < 2; ++l) {
    k_gemm_h128<<<MTILES, 256, 0, stream>>>(xb, wlb + (size_t)l * H * H,
                                            wrb + (size_t)l * H * H, yl, yr);
    agg_combine(b_b + (size_t)l * H, xb);
  }

  // ---- angle branch ----
  k_angle_gbf_gemm<<<MTILES, 256, 0, stream>>>(angle, wla1, wra1, yl, yr);
  agg_combine(b_a1, xa);
  for (int l = 0; l < 2; ++l) {
    k_gemm_h128<<<MTILES, 256, 0, stream>>>(xa, wla + (size_t)l * H * H,
                                            wra + (size_t)l * H * H, yl, yr);
    agg_combine(b_a + (size_t)l * H, xa);
  }

  // ---- pool + MLP head ----
  k_pool<<<N_CRYS, HCAT, 0, stream>>>((const unsigned short*)xb,
                                      (const unsigned short*)xa, cidx, pooled);
  k_linear<<<(N_CRYS * HCAT + 255) / 256, 256, 0, stream>>>(
      pooled, W_mlp, b_mlp, t1, N_CRYS, HCAT, HCAT);
  k_linear<<<(N_CRYS * HCAT + 255) / 256, 256, 0, stream>>>(
      t1, W_mlp + (size_t)HCAT * HCAT, b_mlp + HCAT, t2, N_CRYS, HCAT, HCAT);
  k_linear<<<(N_CRYS * 2 + 255) / 256, 256, 0, stream>>>(
      t2, W_fc, b_fc, (float*)d_out, N_CRYS, HCAT, 2);
}
